// GCN_49916109914532
// MI455X (gfx1250) — matrix operand order repaired and checked by reading, not compiled, not run on hardware
//
#include <hip/hip_runtime.h>

#define NROW   16384
#define NFEAT  512
#define NHID   256
#define NH2    64
#define NCLS   10

typedef __attribute__((ext_vector_type(16))) __bf16 v16bf;
typedef __attribute__((ext_vector_type(8)))  __bf16 v8bf;
typedef __attribute__((ext_vector_type(4)))  __bf16 v4bf;
typedef __attribute__((ext_vector_type(8)))  float  v8f;
typedef __attribute__((ext_vector_type(4)))  float  v4f;

__device__ __forceinline__ v8f wmma_bf16(v16bf a, v16bf b, v8f c) {
  return __builtin_amdgcn_wmma_f32_16x16x32_bf16(
      /*neg_a=*/false, a, /*neg_b=*/false, b,
      /*c_mod=*/(short)0, c, /*reuse_a=*/false, /*reuse_b=*/false);
}

// A-fragment: lane holds row m, 16 contiguous K values [kb..kb+15]
__device__ __forceinline__ v16bf ld_fragA(const __bf16* p) {
  v8bf lo = *(const v8bf*)p;
  v8bf hi = *(const v8bf*)(p + 8);
  v16bf r;
#pragma unroll
  for (int i = 0; i < 8; ++i) { r[i] = lo[i]; r[i + 8] = hi[i]; }
  return r;
}

// B-fragment: lane holds col n, 16 contiguous K values (lanes<16: K0-15, else K16-31)
__device__ __forceinline__ v16bf ld_fragB(const __bf16* p) {
  v8bf lo = *(const v8bf*)p;
  v8bf hi = *(const v8bf*)(p + 8);
  v16bf r;
#pragma unroll
  for (int i = 0; i < 8; ++i) { r[i] = lo[i]; r[i + 8] = hi[i]; }
  return r;
}

__device__ __forceinline__ v8f vzero8() {
  v8f z;
#pragma unroll
  for (int i = 0; i < 8; ++i) z[i] = 0.0f;
  return z;
}

__device__ __forceinline__ void cvt_store4(__bf16* dst, v4f v) {
  v4bf b;
#pragma unroll
  for (int j = 0; j < 4; ++j) b[j] = (__bf16)v[j];
  *(v4bf*)dst = b;
}

// ---- K0: transpose + convert weight matrix [K][Nc] f32 row-major ->
//          [Nc][K] bf16 col-major
__global__ void k_tcvt(__bf16* __restrict__ dst, const float* __restrict__ src,
                       int K, int Nc) {
  int i = blockIdx.x * blockDim.x + threadIdx.x;
  if (i >= K * Nc) return;
  int k = i % K, n = i / K;
  dst[(size_t)n * K + k] = (__bf16)src[(size_t)k * Nc + n];
}

// ---- K1: H1 = x @ W1  -> bf16 col-major [NHID][NROW]
// 64-row block, all 256 cols; double-buffered LDS A tile, pair-unrolled
// software pipeline (no register rotation copies).
__global__ void __launch_bounds__(256) k_xw1(const float* __restrict__ x,
                                             const __bf16* __restrict__ W1bf,
                                             __bf16* __restrict__ H1bf) {
  __shared__ __align__(16) __bf16 As[2][64 * 32];
  const int rowBase = blockIdx.x * 64;
  const int tid  = threadIdx.x;
  const int wave = tid >> 5, lane = tid & 31;
  const int m    = lane & 15;
  const int kbA  = (lane < 16) ? 0 : 16;
  const int kbB  = (lane < 16) ? 0 : 16;
  const int mb   = (lane < 16) ? 0 : 8;
  const int srow = tid >> 3;        // 0..31 (and +32 for second half)
  const int skq  = (tid & 7) << 2;  // 0,4,..,28
  const int so0  = srow * 32 + skq;
  const int so1  = (srow + 32) * 32 + skq;

  const float* gp0 = x + (size_t)(rowBase + srow) * NFEAT + skq;
  const float* gp1 = x + (size_t)(rowBase + srow + 32) * NFEAT + skq;
  const int col0 = (wave * 2) * 16 + m;
  const __bf16* bp0 = W1bf + (size_t)col0 * NFEAT + kbB;
  const __bf16* bp1 = W1bf + (size_t)(col0 + 16) * NFEAT + kbB;

  v8f acc[4][2];
#pragma unroll
  for (int r = 0; r < 4; ++r)
#pragma unroll
    for (int c = 0; c < 2; ++c) acc[r][c] = vzero8();

  v16bf bcur0, bcur1, bn0, bn1;
  // prologue: stage tile 0 into As[0], load B frags for kb=0
  cvt_store4(&As[0][so0], __builtin_nontemporal_load((const v4f*)gp0));
  cvt_store4(&As[0][so1], __builtin_nontemporal_load((const v4f*)gp1));
  bcur0 = ld_fragB(bp0);
  bcur1 = ld_fragB(bp1);
  __syncthreads();

  for (int kb = 0; kb < NFEAT; kb += 64) {
    // ---- body A: compute tile kb from As[0]; stage kb+32 -> As[1]
    {
      v4f s0 = __builtin_nontemporal_load((const v4f*)(gp0 + kb + 32));
      v4f s1 = __builtin_nontemporal_load((const v4f*)(gp1 + kb + 32));
      bn0 = ld_fragB(bp0 + kb + 32);
      bn1 = ld_fragB(bp1 + kb + 32);
      v16bf afr[4];
#pragma unroll
      for (int r = 0; r < 4; ++r)
        afr[r] = ld_fragA(&As[0][(r * 16 + m) * 32 + kbA]);
#pragma unroll
      for (int r = 0; r < 4; ++r) acc[r][0] = wmma_bf16(afr[r], bcur0, acc[r][0]);
#pragma unroll
      for (int r = 0; r < 4; ++r) acc[r][1] = wmma_bf16(afr[r], bcur1, acc[r][1]);
      cvt_store4(&As[1][so0], s0);
      cvt_store4(&As[1][so1], s1);
      __syncthreads();
    }
    // ---- body B: compute tile kb+32 from As[1]; stage kb+64 -> As[0]
    {
      const bool hasNext = (kb + 64) < NFEAT;
      v4f s0, s1;
      if (hasNext) {
        s0 = __builtin_nontemporal_load((const v4f*)(gp0 + kb + 64));
        s1 = __builtin_nontemporal_load((const v4f*)(gp1 + kb + 64));
        bcur0 = ld_fragB(bp0 + kb + 64);
        bcur1 = ld_fragB(bp1 + kb + 64);
      }
      v16bf afr[4];
#pragma unroll
      for (int r = 0; r < 4; ++r)
        afr[r] = ld_fragA(&As[1][(r * 16 + m) * 32 + kbA]);
#pragma unroll
      for (int r = 0; r < 4; ++r) acc[r][0] = wmma_bf16(afr[r], bn0, acc[r][0]);
#pragma unroll
      for (int r = 0; r < 4; ++r) acc[r][1] = wmma_bf16(afr[r], bn1, acc[r][1]);
      if (hasNext) {
        cvt_store4(&As[0][so0], s0);
        cvt_store4(&As[0][so1], s1);
      }
      __syncthreads();
    }
  }

#pragma unroll
  for (int c = 0; c < 2; ++c) {
    int col = (wave * 2 + c) * 16 + m;
#pragma unroll
    for (int r = 0; r < 4; ++r) {
      v8bf d;
#pragma unroll
      for (int i = 0; i < 8; ++i) d[i] = (__bf16)acc[r][c][i];
      *(v8bf*)(H1bf + (size_t)col * NROW + rowBase + r * 16 + mb) = d;
    }
  }
}

// ---- K2: H = relu(adj @ H1 + b1)  -> bf16 row-major [NROW][NHID]
__global__ void __launch_bounds__(256) k_adj_h1(const float* __restrict__ adj,
                                                const __bf16* __restrict__ H1bf,
                                                const float* __restrict__ b1,
                                                __bf16* __restrict__ Hbf) {
  __shared__ __align__(16) __bf16 As[2][64 * 32];
  const int rowBase = blockIdx.x * 64;
  const int tid  = threadIdx.x;
  const int wave = tid >> 5, lane = tid & 31;
  const int m    = lane & 15;
  const int kbA  = (lane < 16) ? 0 : 16;
  const int kbB  = (lane < 16) ? 0 : 16;
  const int mb   = (lane < 16) ? 0 : 8;
  const int srow = tid >> 3;
  const int skq  = (tid & 7) << 2;
  const int so0  = srow * 32 + skq;
  const int so1  = (srow + 32) * 32 + skq;

  const float* gp0 = adj + (size_t)(rowBase + srow) * NROW + skq;
  const float* gp1 = adj + (size_t)(rowBase + srow + 32) * NROW + skq;
  const int col0 = (wave * 2) * 16 + m;
  const __bf16* bp0 = H1bf + (size_t)col0 * NROW + kbB;
  const __bf16* bp1 = H1bf + (size_t)(col0 + 16) * NROW + kbB;

  v8f acc[4][2];
#pragma unroll
  for (int r = 0; r < 4; ++r)
#pragma unroll
    for (int c = 0; c < 2; ++c) acc[r][c] = vzero8();

  v16bf bcur0, bcur1, bn0, bn1;
  cvt_store4(&As[0][so0], __builtin_nontemporal_load((const v4f*)gp0));
  cvt_store4(&As[0][so1], __builtin_nontemporal_load((const v4f*)gp1));
  bcur0 = ld_fragB(bp0);
  bcur1 = ld_fragB(bp1);
  __syncthreads();

  for (int kb = 0; kb < NROW; kb += 64) {
    // ---- body A: compute tile kb from As[0]; stage kb+32 -> As[1]
    {
      v4f s0 = __builtin_nontemporal_load((const v4f*)(gp0 + kb + 32));
      v4f s1 = __builtin_nontemporal_load((const v4f*)(gp1 + kb + 32));
      __builtin_prefetch(gp0 + kb + 64, 0, 0);
      bn0 = ld_fragB(bp0 + kb + 32);
      bn1 = ld_fragB(bp1 + kb + 32);
      v16bf afr[4];
#pragma unroll
      for (int r = 0; r < 4; ++r)
        afr[r] = ld_fragA(&As[0][(r * 16 + m) * 32 + kbA]);
#pragma unroll
      for (int r = 0; r < 4; ++r) acc[r][0] = wmma_bf16(afr[r], bcur0, acc[r][0]);
#pragma unroll
      for (int r = 0; r < 4; ++r) acc[r][1] = wmma_bf16(afr[r], bcur1, acc[r][1]);
      cvt_store4(&As[1][so0], s0);
      cvt_store4(&As[1][so1], s1);
      __syncthreads();
    }
    // ---- body B: compute tile kb+32 from As[1]; stage kb+64 -> As[0]
    {
      const bool hasNext = (kb + 64) < NROW;
      v4f s0, s1;
      if (hasNext) {
        s0 = __builtin_nontemporal_load((const v4f*)(gp0 + kb + 64));
        s1 = __builtin_nontemporal_load((const v4f*)(gp1 + kb + 64));
        __builtin_prefetch(gp0 + kb + 96, 0, 0);
        bcur0 = ld_fragB(bp0 + kb + 64);
        bcur1 = ld_fragB(bp1 + kb + 64);
      }
      v16bf afr[4];
#pragma unroll
      for (int r = 0; r < 4; ++r)
        afr[r] = ld_fragA(&As[1][(r * 16 + m) * 32 + kbA]);
#pragma unroll
      for (int r = 0; r < 4; ++r) acc[r][0] = wmma_bf16(afr[r], bn0, acc[r][0]);
#pragma unroll
      for (int r = 0; r < 4; ++r) acc[r][1] = wmma_bf16(afr[r], bn1, acc[r][1]);
      if (hasNext) {
        cvt_store4(&As[0][so0], s0);
        cvt_store4(&As[0][so1], s1);
      }
      __syncthreads();
    }
  }

#pragma unroll
  for (int c = 0; c < 2; ++c) {
    int col = (wave * 2 + c) * 16 + m;
    float bias = b1[col];
#pragma unroll
    for (int r = 0; r < 4; ++r) {
      int mrow = rowBase + r * 16 + mb;
#pragma unroll
      for (int i = 0; i < 8; ++i) {
        float v = fmaxf(acc[r][c][i] + bias, 0.0f);
        Hbf[(size_t)(mrow + i) * NHID + col] = (__bf16)v;
      }
    }
  }
}

// ---- K3: H2 = H @ W2  -> bf16 col-major [NH2][NROW]
__global__ void __launch_bounds__(256) k_hw2(const __bf16* __restrict__ Hbf,
                                             const __bf16* __restrict__ W2bf,
                                             __bf16* __restrict__ H2bf) {
  const int rowBase = blockIdx.x * 64;
  const int tid  = threadIdx.x;
  const int wave = tid >> 5, lane = tid & 31;
  const int m    = lane & 15;
  const int kbA  = (lane < 16) ? 0 : 16;
  const int kbB  = (lane < 16) ? 0 : 16;
  const int mb   = (lane < 16) ? 0 : 8;
  const int rt   = wave >> 1;
  const int cb   = (wave & 1) * 2;

  v8f acc[2];
  acc[0] = vzero8(); acc[1] = vzero8();

#pragma unroll
  for (int kb = 0; kb < NHID; kb += 32) {
    v16bf a = ld_fragA(Hbf + (size_t)(rowBase + rt * 16 + m) * NHID + kb + kbA);
#pragma unroll
    for (int c = 0; c < 2; ++c) {
      int col = (cb + c) * 16 + m;
      v16bf bfr = ld_fragB(W2bf + (size_t)col * NHID + kb + kbB);
      acc[c] = wmma_bf16(a, bfr, acc[c]);
    }
  }
#pragma unroll
  for (int c = 0; c < 2; ++c) {
    int col = (cb + c) * 16 + m;
    v8bf d;
#pragma unroll
    for (int i = 0; i < 8; ++i) d[i] = (__bf16)acc[c][i];
    *(v8bf*)(H2bf + (size_t)col * NROW + rowBase + rt * 16 + mb) = d;
  }
}

// ---- K4: H3 = adj @ H2 + b2  -> f32 row-major [NROW][NH2]
__global__ void __launch_bounds__(256) k_adj_h2(const float* __restrict__ adj,
                                                const __bf16* __restrict__ H2bf,
                                                const float* __restrict__ b2,
                                                float* __restrict__ H3) {
  __shared__ __align__(16) __bf16 As[2][64 * 32];
  const int rowBase = blockIdx.x * 64;
  const int tid  = threadIdx.x;
  const int wave = tid >> 5, lane = tid & 31;
  const int m    = lane & 15;
  const int kbA  = (lane < 16) ? 0 : 16;
  const int kbB  = (lane < 16) ? 0 : 16;
  const int mb   = (lane < 16) ? 0 : 8;
  const int rt   = wave >> 1;
  const int cb   = (wave & 1) * 2;
  const int srow = tid >> 3;
  const int skq  = (tid & 7) << 2;
  const int so0  = srow * 32 + skq;
  const int so1  = (srow + 32) * 32 + skq;

  const float* gp0 = adj + (size_t)(rowBase + srow) * NROW + skq;
  const float* gp1 = adj + (size_t)(rowBase + srow + 32) * NROW + skq;
  const int col0 = (cb + 0) * 16 + m;
  const __bf16* bp0 = H2bf + (size_t)col0 * NROW + kbB;
  const __bf16* bp1 = H2bf + (size_t)(col0 + 16) * NROW + kbB;

  v8f acc[2];
  acc[0] = vzero8(); acc[1] = vzero8();

  v16bf bcur0, bcur1, bn0, bn1;
  cvt_store4(&As[0][so0], __builtin_nontemporal_load((const v4f*)gp0));
  cvt_store4(&As[0][so1], __builtin_nontemporal_load((const v4f*)gp1));
  bcur0 = ld_fragB(bp0);
  bcur1 = ld_fragB(bp1);
  __syncthreads();

  for (int kb = 0; kb < NROW; kb += 64) {
    // ---- body A
    {
      v4f s0 = __builtin_nontemporal_load((const v4f*)(gp0 + kb + 32));
      v4f s1 = __builtin_nontemporal_load((const v4f*)(gp1 + kb + 32));
      __builtin_prefetch(gp0 + kb + 64, 0, 0);
      bn0 = ld_fragB(bp0 + kb + 32);
      bn1 = ld_fragB(bp1 + kb + 32);
      v16bf a = ld_fragA(&As[0][(rt * 16 + m) * 32 + kbA]);
      acc[0] = wmma_bf16(a, bcur0, acc[0]);
      acc[1] = wmma_bf16(a, bcur1, acc[1]);
      cvt_store4(&As[1][so0], s0);
      cvt_store4(&As[1][so1], s1);
      __syncthreads();
    }
    // ---- body B
    {
      const bool hasNext = (kb + 64) < NROW;
      v4f s0, s1;
      if (hasNext) {
        s0 = __builtin_nontemporal_load((const v4f*)(gp0 + kb + 64));
        s1 = __builtin_nontemporal_load((const v4f*)(gp1 + kb + 64));
        __builtin_prefetch(gp0 + kb + 96, 0, 0);
        bcur0 = ld_fragB(bp0 + kb + 64);
        bcur1 = ld_fragB(bp1 + kb + 64);
      }
      v16bf a = ld_fragA(&As[1][(rt * 16 + m) * 32 + kbA]);
      acc[0] = wmma_bf16(a, bn0, acc[0]);
      acc[1] = wmma_bf16(a, bn1, acc[1]);
      if (hasNext) {
        cvt_store4(&As[0][so0], s0);
        cvt_store4(&As[0][so1], s1);
      }
      __syncthreads();
    }
  }

#pragma unroll
  for (int c = 0; c < 2; ++c) {
    int col = (cb + c) * 16 + m;
    float bias = b2[col];
    int mrow = rowBase + rt * 16 + mb;
#pragma unroll
    for (int i = 0; i < 8; ++i)
      H3[(size_t)(mrow + i) * NH2 + col] = acc[c][i] + bias;
  }
}

// ---- K5: g = columnwise max over H3
__global__ void __launch_bounds__(256) k_colmax(const float* __restrict__ H3,
                                                float* __restrict__ g) {
  __shared__ float red[256];
  const int col = blockIdx.x;
  float mval = -3.402823466e38f;
  for (int r = threadIdx.x; r < NROW; r += 256)
    mval = fmaxf(mval, H3[(size_t)r * NH2 + col]);
  red[threadIdx.x] = mval;
  __syncthreads();
  for (int s = 128; s > 0; s >>= 1) {
    if (threadIdx.x < s)
      red[threadIdx.x] = fmaxf(red[threadIdx.x], red[threadIdx.x + s]);
    __syncthreads();
  }
  if (threadIdx.x == 0) g[col] = red[0];
}

// ---- K6: tiny MLP head
__global__ void k_mlp(const float* __restrict__ g,
                      const float* __restrict__ W3, const float* __restrict__ b3,
                      const float* __restrict__ W4, const float* __restrict__ b4,
                      const float* __restrict__ W5, const float* __restrict__ b5,
                      float* __restrict__ out) {
  __shared__ float s3[32], s4[16];
  const int t = threadIdx.x;
  if (t < 32) {
    float s = b3[t];
    for (int k = 0; k < 64; ++k) s += g[k] * W3[k * 32 + t];
    s3[t] = fmaxf(s, 0.0f);
  }
  __syncthreads();
  if (t < 16) {
    float s = b4[t];
    for (int k = 0; k < 32; ++k) s += s3[k] * W4[k * 16 + t];
    s4[t] = fmaxf(s, 0.0f);
  }
  __syncthreads();
  if (t < NCLS) {
    float s = b5[t];
    for (int k = 0; k < 16; ++k) s += s4[k] * W5[k * NCLS + t];
    out[t] = s;
  }
}

extern "C" void kernel_launch(void* const* d_in, const int* in_sizes, int n_in,
                              void* d_out, int out_size, void* d_ws, size_t ws_size,
                              hipStream_t stream) {
  const float* x   = (const float*)d_in[0];
  const float* adj = (const float*)d_in[1];
  const float* W1  = (const float*)d_in[2];
  const float* b1  = (const float*)d_in[3];
  const float* W2  = (const float*)d_in[4];
  const float* b2  = (const float*)d_in[5];
  const float* W3  = (const float*)d_in[6];
  const float* b3  = (const float*)d_in[7];
  const float* W4  = (const float*)d_in[8];
  const float* b4  = (const float*)d_in[9];
  const float* W5  = (const float*)d_in[10];
  const float* b5  = (const float*)d_in[11];
  float* out = (float*)d_out;

  char* ws = (char*)d_ws;
  // ws layout (bytes):
  //   [0,            262144)   W1bf  bf16 col-major [256][512]
  //   [262144,       294912)   W2bf  bf16 col-major [64][256]
  //   [524288,      8912896)   H1bf  bf16 col-major [256][16384]
  //   [8912896,    17301504)   Hbf   bf16 row-major [16384][256]
  //   [17301504,   19398656)   H2bf  bf16 col-major [64][16384]
  //   [19398656,   23592960)   H3    f32  row-major [16384][64]
  //   [23592960,   23593216)   g     f32  [64]
  __bf16* W1bf = (__bf16*)(ws + 0);
  __bf16* W2bf = (__bf16*)(ws + 262144);
  __bf16* H1bf = (__bf16*)(ws + 524288);
  __bf16* Hbf  = (__bf16*)(ws + 8912896);
  __bf16* H2bf = (__bf16*)(ws + 17301504);
  float*  H3   = (float*) (ws + 19398656);
  float*  g    = (float*) (ws + 23592960);

  k_tcvt<<<(NFEAT * NHID + 255) / 256, 256, 0, stream>>>(W1bf, W1, NFEAT, NHID);
  k_tcvt<<<(NHID * NH2 + 255) / 256, 256, 0, stream>>>(W2bf, W2, NHID, NH2);
  k_xw1   <<<NROW / 64, 256, 0, stream>>>(x, W1bf, H1bf);
  k_adj_h1<<<NROW / 64, 256, 0, stream>>>(adj, H1bf, b1, Hbf);
  k_hw2   <<<NROW / 64, 256, 0, stream>>>(Hbf, W2bf, H2bf);
  k_adj_h2<<<NROW / 64, 256, 0, stream>>>(adj, H2bf, b2, H3);
  k_colmax<<<NH2, 256, 0, stream>>>(H3, g);
  k_mlp   <<<1, 32, 0, stream>>>(g, W3, b3, W4, b4, W5, b5, out);
}
